// ArchetypalNeuralMemory_3607772529028
// MI455X (gfx1250) — matrix operand order repaired and checked by reading, not compiled, not run on hardware
//
#include <hip/hip_runtime.h>

typedef __attribute__((ext_vector_type(16))) _Float16 v16h;
typedef __attribute__((ext_vector_type(8)))  float    v8f;
using u16 = unsigned short;
using u32 = unsigned int;

#define D512 512
#define TOK  16384   // B*S = 4*4096
#define BM 128
#define BN 64
#define BK 32
#define LDSK 40      // BK + 8 halves pad (80B rows, 16B aligned)

__device__ __forceinline__ u16 f2h(float x){ union{_Float16 h; u16 u;}c; c.h=(_Float16)x; return c.u; }
__device__ __forceinline__ float h2f(u16 u){ union{_Float16 h; u16 u;}c; c.u=u; return (float)c.h; }
__device__ __forceinline__ float sigm(float x){ return 1.f/(1.f+__expf(-x)); }
__device__ __forceinline__ float siluf(float x){ return x*sigm(x); }

union Frag { v16h v; uint4 q[2]; };

enum { EPI_F32=0, EPI_SILU_F32=1, EPI_SILU_F16=2, EPI_F16=3, EPI_HZ=4,
       EPI_SCANOUT=5, EPI_DPRE=6, EPI_UPDATE=7 };

// ---------------------------------------------------------------------------
// Generic f16-in / f32-accum WMMA GEMM: C[M,N] = A[M,K] * B
//   TRANSA: A physically stored [K, M] (row-major), read transposed
//   TRANSB: B is W row-major [N, K]  (i.e. computes A @ W^T); else B is [K, N]
//   rowsPerBatch>0: B pointer advanced by bStrideB per `rowsPerBatch` rows of A
// LDS layouts: sA[m][k], sB[n][k]  -> every WMMA fragment = two b128 DS loads
// ---------------------------------------------------------------------------
template<bool TRANSA, bool TRANSB, int EPI>
__global__ __launch_bounds__(256)
void gemm_f16(const u16* __restrict__ A, int lda,
              const u16* __restrict__ B, int ldb,
              int rowsPerBatch, long bStrideB,
              int M, int N, int K,
              float* __restrict__ outF, u16* __restrict__ outH,
              const u16* __restrict__ vsrc, u16* __restrict__ ebuf,
              const float* __restrict__ zbuf,
              float* __restrict__ wbuf, float* __restrict__ mbuf,
              const float* __restrict__ gates, int step, float escale)
{
  __shared__ __align__(16) u16 sA[BM*LDSK];
  __shared__ __align__(16) u16 sB[BN*LDSK];
  const int tid = threadIdx.x;
  const int bm = blockIdx.y*BM, bn = blockIdx.x*BN;
  const u16* Bp = B;
  if (rowsPerBatch > 0) Bp += (long)(bm / rowsPerBatch) * bStrideB;

  const int wid = tid>>5, lane = tid&31, lo = lane&15, hi = lane>>4;
  const int wm = (wid&3)*32, wn = (wid>>2)*32;

  const v8f vzero = {0.f,0.f,0.f,0.f,0.f,0.f,0.f,0.f};
  v8f acc[2][2];
  #pragma unroll
  for(int i=0;i<2;i++){ acc[i][0]=vzero; acc[i][1]=vzero; }

  for (int kk=0; kk<K; kk+=BK){
    // ---- stage A tile: sA[m][k] ----
    if (!TRANSA){
      const int r = tid>>1, c = (tid&1)*16;
      const u16* src = A + (long)(bm+r)*lda + kk + c;
      uint4 d0 = *(const uint4*)src;
      uint4 d1 = *(const uint4*)(src+8);
      if (kk+BK < K) __builtin_prefetch(src+BK, 0, 3);
      *(uint4*)&sA[r*LDSK + c]     = d0;
      *(uint4*)&sA[r*LDSK + c + 8] = d1;
    } else {
      const int k = tid>>3, mb = (tid&7)*16;
      const u16* src = A + (long)(kk+k)*lda + bm + mb;
      uint4 d0 = *(const uint4*)src;
      uint4 d1 = *(const uint4*)(src+8);
      if (kk+BK < K) __builtin_prefetch(src + (long)BK*lda, 0, 3);
      const u16* s0=(const u16*)&d0; const u16* s1=(const u16*)&d1;
      #pragma unroll
      for(int i=0;i<8;i++){ sA[(mb+i)*LDSK+k]=s0[i]; sA[(mb+8+i)*LDSK+k]=s1[i]; }
    }
    // ---- stage B tile: sB[n][k] ----
    if (TRANSB){
      const int n = tid>>2, kb = (tid&3)*8;
      const u16* src = Bp + (long)(bn+n)*ldb + kk + kb;
      if (kk+BK < K) __builtin_prefetch(src+BK, 0, 3);
      *(uint4*)&sB[n*LDSK+kb] = *(const uint4*)src;
    } else {
      const int k = tid>>3, nb = (tid&7)*8;
      const u16* src = Bp + (long)(kk+k)*ldb + bn + nb;
      uint4 d = *(const uint4*)src;
      const u16* s=(const u16*)&d;
      #pragma unroll
      for(int i=0;i<8;i++) sB[(nb+i)*LDSK+k]=s[i];
    }
    __syncthreads();

    // ---- fragments per ISA lane layout (16-bit A 16x32, B 32x16) ----
    Frag a[2], b[2];
    #pragma unroll
    for(int s2=0;s2<2;s2++){
      const int abase = (wm+s2*16+lo)*LDSK + hi*8;   // elems 0..7 -> k=hi*8.., 8..15 -> k=hi*8+16..
      a[s2].q[0] = *(const uint4*)&sA[abase];
      a[s2].q[1] = *(const uint4*)&sA[abase+16];
      const int bbase = (wn+s2*16+lo)*LDSK + hi*8;   // elems 0..7 -> k=hi*8.., 8..15 -> k=hi*8+16..
      b[s2].q[0] = *(const uint4*)&sB[bbase];
      b[s2].q[1] = *(const uint4*)&sB[bbase+16];
    }
    #pragma unroll
    for(int i=0;i<2;i++)
      #pragma unroll
      for(int j=0;j<2;j++)
        acc[i][j] = __builtin_amdgcn_wmma_f32_16x16x32_f16(
            false, a[i].v, false, b[j].v, (short)0, acc[i][j], false, false);
    __syncthreads();
  }

  // ---- epilogue (C layout: vgpr r -> row m0+r+hi*8, lane lo -> col) ----
  #pragma unroll
  for(int i=0;i<2;i++){
    #pragma unroll
    for(int j=0;j<2;j++){
      const int row0 = bm + wm + i*16 + hi*8;
      const int col  = bn + wn + j*16 + lo;
      #pragma unroll
      for(int r=0;r<8;r++){
        const int row = row0 + r;
        const float val = acc[i][j][r];
        const long idx = (long)row*N + col;
        if constexpr (EPI==EPI_F32)           outF[idx] = val;
        else if constexpr (EPI==EPI_SILU_F32) outF[idx] = siluf(val);
        else if constexpr (EPI==EPI_SILU_F16) outH[idx] = f2h(siluf(val));
        else if constexpr (EPI==EPI_F16)      outH[idx] = f2h(val);
        else if constexpr (EPI==EPI_HZ){      outF[idx] = val; outH[idx] = f2h(siluf(val)); }
        else if constexpr (EPI==EPI_SCANOUT){
          if (row < 256){                       // retrieved: scatter back to [B,S,D]
            long grow = (long)(row>>6)*4096 + (long)step*64 + (row&63);
            outH[grow*D512 + col] = f2h(val);
          } else {                              // MSE error e = (pred - v) * 2/(B*C*D)
            int rr = row - 256;
            long grow = (long)(rr>>6)*4096 + (long)step*64 + (rr&63);
            float vv = h2f(vsrc[grow*D512 + col]);
            ebuf[(long)rr*D512 + col] = f2h((val - vv)*escale);
          }
        }
        else if constexpr (EPI==EPI_DPRE){      // dh * silu'(z)
          float z = zbuf[idx];
          float s = sigm(z);
          outH[idx] = f2h(val * (s*(1.f + z*(1.f-s))));
        }
        else if constexpr (EPI==EPI_UPDATE){    // mom = eta*mom - theta*g; W = (1-a)*W + mom
          float aG = gates[step], th = gates[64+step], et = gates[128+step];
          float mmv = et*mbuf[idx] - th*val;
          mbuf[idx] = mmv;
          wbuf[idx] = (1.f - aG)*wbuf[idx] + mmv;
        }
      }
    }
  }
}

// ---------------------------------------------------------------------------
// Small support kernels
// ---------------------------------------------------------------------------
__device__ __forceinline__ float blockReduceSum(float v, float* red){
  const int tid = threadIdx.x;
  red[tid] = v; __syncthreads();
  #pragma unroll
  for(int s=128; s>0; s>>=1){ if (tid < s) red[tid] += red[tid+s]; __syncthreads(); }
  float r = red[0]; __syncthreads();
  return r;
}

__global__ __launch_bounds__(256)
void init_kernel(const float* __restrict__ memW, float* __restrict__ wf,
                 float* __restrict__ mf, float* __restrict__ gates){
  int i = blockIdx.x*256 + threadIdx.x;
  if (i < 2*D512*D512){ wf[i] = memW[i]; mf[i] = 0.f; }
  if (i < 192) gates[i] = 0.f;
}

__global__ __launch_bounds__(256)
void cvt_f32_f16(const float* __restrict__ in, u16* __restrict__ out, int n){
  int i = blockIdx.x*256 + threadIdx.x;
  if (i < n) out[i] = f2h(in[i]);
}

// rmsnorm with two gains -> x_store(f16) and x_retr(f16)
__global__ __launch_bounds__(256)
void rms_kernel(const float* __restrict__ x, const float* __restrict__ gs,
                const float* __restrict__ gr, u16* __restrict__ xs, u16* __restrict__ xr){
  __shared__ float red[256];
  const long row = blockIdx.x; const int tid = threadIdx.x;
  const float* xp = x + row*D512;
  float v0 = xp[tid], v1 = xp[tid+256];
  float s = blockReduceSum(v0*v0 + v1*v1, red);
  float sc = rsqrtf(s*(1.f/512.f) + 1.1920929e-07f);
  xs[row*D512+tid]     = f2h(v0*sc*gs[tid]);
  xs[row*D512+tid+256] = f2h(v1*sc*gs[tid+256]);
  xr[row*D512+tid]     = f2h(v0*sc*gr[tid]);
  xr[row*D512+tid+256] = f2h(v1*sc*gr[tid+256]);
}

// l2 normalize row of f32, emit f16
__global__ __launch_bounds__(256)
void l2n_kernel(const float* __restrict__ in, u16* __restrict__ out){
  __shared__ float red[256];
  const long row = blockIdx.x; const int tid = threadIdx.x;
  float v0 = in[row*D512+tid], v1 = in[row*D512+tid+256];
  float s = blockReduceSum(v0*v0 + v1*v1, red);
  float inv = 1.f / fmaxf(sqrtf(s), 1e-12f);
  out[row*D512+tid]     = f2h(v0*inv);
  out[row*D512+tid+256] = f2h(v1*inv);
}

// chunk-mean gates: gates[t]=alpha, gates[64+t]=theta, gates[128+t]=eta
__global__ __launch_bounds__(256)
void gates_kernel(const float* __restrict__ x,
                  const float* __restrict__ Wgd, const float* __restrict__ bgd,
                  const float* __restrict__ Wgl, const float* __restrict__ bgl,
                  const float* __restrict__ Wgm, const float* __restrict__ bgm,
                  float* __restrict__ gates){
  __shared__ float cm[D512];
  __shared__ float red[256];
  const int t = blockIdx.x, tid = threadIdx.x;
  float a0=0.f, a1=0.f, a2=0.f;
  for (int b=0;b<4;b++){
    __syncthreads();
    for (int d=tid; d<D512; d+=256){
      float s=0.f;
      const float* xp = x + ((long)b*4096 + (long)t*64)*D512 + d;
      for (int c=0;c<64;c++) s += xp[(long)c*D512];
      cm[d] = s*(1.f/64.f);
    }
    __syncthreads();
    for (int j=tid; j<D512; j+=256){
      float d0=bgd[j], d1=bgl[j], d2=bgm[j];
      const float* w0 = Wgd + (long)j*D512;
      const float* w1 = Wgl + (long)j*D512;
      const float* w2 = Wgm + (long)j*D512;
      for (int d=0; d<D512; d++){
        float c = cm[d];
        d0 += c*w0[d]; d1 += c*w1[d]; d2 += c*w2[d];
      }
      a0 += sigm(d0); a1 += sigm(d1); a2 += sigm(d2);
    }
  }
  float s0 = blockReduceSum(a0, red);
  float s1 = blockReduceSum(a1, red);
  float s2 = blockReduceSum(a2, red);
  if (tid==0){
    gates[t]      = s0*(1.f/512.f)*0.01f*0.25f;  // DEC, /B
    gates[64+t]   = s1*(1.f/512.f)*0.1f *0.25f;  // LR
    gates[128+t]  = s2*(1.f/512.f)*0.9f *0.25f;  // MOM
  }
}

// gather chunk t of q,k into contiguous [256,512] f16 buffers (uint copies)
__global__ __launch_bounds__(256)
void gather_qk(const u32* __restrict__ Q, const u32* __restrict__ Kk,
               u32* __restrict__ qt, u32* __restrict__ kt, int t){
  int gid = blockIdx.x*256 + threadIdx.x;      // 65536 = 256 rows * 256 uints
  int r = gid>>8, cu = gid&255;
  long grow = (long)(r>>6)*4096 + (long)t*64 + (r&63);
  qt[gid] = Q[grow*256 + cu];
  kt[gid] = Kk[grow*256 + cu];
}

// ---------------------------------------------------------------------------
extern "C" void kernel_launch(void* const* d_in, const int* in_sizes, int n_in,
                              void* d_out, int out_size, void* d_ws, size_t ws_size,
                              hipStream_t stream) {
  const float* x    = (const float*)d_in[0];
  const float* Mmat = (const float*)d_in[1];
  const float* memW = (const float*)d_in[2];
  const float* Wk   = (const float*)d_in[3];
  const float* Wv   = (const float*)d_in[4];
  const float* Wq   = (const float*)d_in[5];
  const float* Wout = (const float*)d_in[6];
  const float* Wgd  = (const float*)d_in[7];
  const float* bgd  = (const float*)d_in[8];
  const float* Wgl  = (const float*)d_in[9];
  const float* bgl  = (const float*)d_in[10];
  const float* Wgm  = (const float*)d_in[11];
  const float* bgm  = (const float*)d_in[12];
  const float* gs   = (const float*)d_in[13];
  const float* gr   = (const float*)d_in[14];
  float* out = (float*)d_out;

  char* wsp = (char*)d_ws; size_t off = 0;
  auto alloc = [&](size_t b)->char*{
    char* p = wsp + off; off += (b + 255) & ~(size_t)255; return p;
  };
  u16*   Q16  = (u16*)  alloc((size_t)TOK*D512*2);
  u16*   K16  = (u16*)  alloc((size_t)TOK*D512*2);
  u16*   V16  = (u16*)  alloc((size_t)TOK*D512*2);
  u16*   XS16 = (u16*)  alloc((size_t)TOK*D512*2);
  u16*   XR16 = (u16*)  alloc((size_t)TOK*D512*2);   // reused as TMP after q-path
  u16*   RET16= (u16*)  alloc((size_t)TOK*D512*2);
  float* PREF = (float*)alloc((size_t)TOK*D512*4);
  float* W0F  = (float*)alloc((size_t)2*D512*D512*4); // W0F,W1F contiguous
  float* M0F  = (float*)alloc((size_t)2*D512*D512*4); // M0F,M1F contiguous
  u16*   W016 = (u16*)  alloc((size_t)2*D512*D512*2); // W016,W116 contiguous
  float* ZF   = (float*)alloc((size_t)512*D512*4);
  u16*   H16  = (u16*)  alloc((size_t)512*D512*2);
  u16*   QT16 = (u16*)  alloc((size_t)2*256*D512*2);  // QT,KT contiguous
  u16*   E16  = (u16*)  alloc((size_t)256*D512*2);
  u16*   DP16 = (u16*)  alloc((size_t)256*D512*2);
  u16*   MB16 = (u16*)  alloc((size_t)4*D512*D512*2);
  u16*   WK16 = (u16*)  alloc((size_t)D512*D512*2);
  u16*   WV16 = (u16*)  alloc((size_t)D512*D512*2);
  u16*   WQ16 = (u16*)  alloc((size_t)D512*D512*2);
  u16*   WO16 = (u16*)  alloc((size_t)D512*D512*2);
  float* GATES= (float*)alloc(192*4);
  float* W1F  = W0F + D512*D512;
  float* M1F  = M0F + D512*D512;
  u16*   W116 = W016 + D512*D512;
  u16*   TMP16= XR16;
  u16*   KT16 = QT16 + 256*D512;
  u16*   HK16 = H16  + 256*D512;
  float* ZK   = ZF   + 256*D512;

  const dim3 gBig(D512/BN, TOK/BM);   // 8 x 128
  const dim3 gW  (D512/BN, D512/BM);  // 8 x 4
  const dim3 gT  (D512/BN, 256/BM);   // 8 x 2
  const float escale = 2.f/131072.f;  // 2/(B*C*D)

  // ---- prologue ----
  init_kernel<<<2048,256,0,stream>>>(memW, W0F, M0F, GATES);
  gates_kernel<<<64,256,0,stream>>>(x, Wgd,bgd, Wgl,bgl, Wgm,bgm, GATES);
  cvt_f32_f16<<<4096,256,0,stream>>>(Mmat, MB16, 4*D512*D512);
  cvt_f32_f16<<<1024,256,0,stream>>>(Wk,   WK16, D512*D512);
  cvt_f32_f16<<<1024,256,0,stream>>>(Wv,   WV16, D512*D512);
  cvt_f32_f16<<<1024,256,0,stream>>>(Wq,   WQ16, D512*D512);
  cvt_f32_f16<<<1024,256,0,stream>>>(Wout, WO16, D512*D512);
  rms_kernel<<<TOK,256,0,stream>>>(x, gs, gr, XS16, XR16);

  // q = l2norm(silu(xr @ Wq^T))
  gemm_f16<false,true,EPI_SILU_F32><<<gBig,256,0,stream>>>(XR16,D512, WQ16,D512, 0,0,
      TOK,D512,D512, PREF,nullptr, nullptr,nullptr,nullptr, nullptr,nullptr, nullptr,0,0.f);
  l2n_kernel<<<TOK,256,0,stream>>>(PREF, Q16);
  // tmp = xs @ M[b]   (per-batch B, no transpose)  -> reuses XR16
  gemm_f16<false,false,EPI_F16><<<gBig,256,0,stream>>>(XS16,D512, MB16,D512, 4096,(long)D512*D512,
      TOK,D512,D512, nullptr,TMP16, nullptr,nullptr,nullptr, nullptr,nullptr, nullptr,0,0.f);
  // k = l2norm(silu(tmp @ Wk^T))
  gemm_f16<false,true,EPI_SILU_F32><<<gBig,256,0,stream>>>(TMP16,D512, WK16,D512, 0,0,
      TOK,D512,D512, PREF,nullptr, nullptr,nullptr,nullptr, nullptr,nullptr, nullptr,0,0.f);
  l2n_kernel<<<TOK,256,0,stream>>>(PREF, K16);
  // v = silu(xs @ Wv^T)
  gemm_f16<false,true,EPI_SILU_F16><<<gBig,256,0,stream>>>(XS16,D512, WV16,D512, 0,0,
      TOK,D512,D512, nullptr,V16, nullptr,nullptr,nullptr, nullptr,nullptr, nullptr,0,0.f);

  // ---- sequential scan over 64 chunks ----
  for (int t=0; t<64; ++t){
    cvt_f32_f16<<<2048,256,0,stream>>>(W0F, W016, 2*D512*D512);       // f16 snapshot of W0,W1
    gather_qk<<<256,256,0,stream>>>((const u32*)Q16,(const u32*)K16,(u32*)QT16,(u32*)KT16,t);
    // [q_t;k_t] @ W0^T -> z (f32, all 512 rows), h = silu(z) (f16)
    gemm_f16<false,true,EPI_HZ><<<gW,256,0,stream>>>(QT16,D512, W016,D512, 0,0,
        512,D512,D512, ZF,H16, nullptr,nullptr,nullptr, nullptr,nullptr, nullptr,0,0.f);
    // h @ W1^T -> retrieved (rows 0..255, scattered) and e = (pred - v_t)*scale (rows 256..511)
    gemm_f16<false,true,EPI_SCANOUT><<<gW,256,0,stream>>>(H16,D512, W116,D512, 0,0,
        512,D512,D512, nullptr,RET16, V16,E16,nullptr, nullptr,nullptr, nullptr,t,escale);
    // gW1 = e^T @ hk ; fused momentum/decay update of W1F (uses pre-update f16 snapshot elsewhere)
    gemm_f16<true,false,EPI_UPDATE><<<gW,256,0,stream>>>(E16,D512, HK16,D512, 0,0,
        512,D512,256, nullptr,nullptr, nullptr,nullptr,nullptr, W1F,M1F, GATES,t,0.f);
    // dpre = (e @ W1_snapshot) * silu'(z_k)
    gemm_f16<false,false,EPI_DPRE><<<gT,256,0,stream>>>(E16,D512, W116,D512, 0,0,
        256,D512,D512, nullptr,DP16, nullptr,nullptr,ZK, nullptr,nullptr, nullptr,0,0.f);
    // gW0 = dpre^T @ k_t ; fused update of W0F
    gemm_f16<true,false,EPI_UPDATE><<<gW,256,0,stream>>>(DP16,D512, KT16,D512, 0,0,
        512,D512,256, nullptr,nullptr, nullptr,nullptr,nullptr, W0F,M0F, GATES,t,0.f);
  }

  // ---- out = retrieved @ Wout^T ----
  gemm_f16<false,true,EPI_F32><<<gBig,256,0,stream>>>(RET16,D512, WO16,D512, 0,0,
      TOK,D512,D512, out,nullptr, nullptr,nullptr,nullptr, nullptr,nullptr, nullptr,0,0.f);

  (void)in_sizes; (void)n_in; (void)out_size; (void)ws_size;
}
